// MultiHeadCfC_34763465294531
// MI455X (gfx1250) — matrix operand order repaired and checked by reading, not compiled, not run on hardware
//
#include <hip/hip_runtime.h>
#include <hip/hip_bf16.h>
#include <math.h>

// dims: V=8, B=256, T=128, U=128, BB=128, H=1024, C=10
typedef __attribute__((ext_vector_type(16))) _Float16 v16h;
typedef __attribute__((ext_vector_type(8)))  _Float16 v8h;
typedef __attribute__((ext_vector_type(8)))  float    v8f;

// ---------------------------------------------------------------------------
// WMMA helpers (CDNA5 wave32, 16x16x32 f16 -> f32)
// ---------------------------------------------------------------------------
__device__ __forceinline__ v8f wmma32(v16h a, v16h b, v8f c) {
  return __builtin_amdgcn_wmma_f32_16x16x32_f16(
      /*neg_a=*/false, a, /*neg_b=*/false, b,
      /*c_mod=*/(short)0, c, /*reuse_a=*/false, /*reuse_b=*/false);
}

// Branch-free activations (inputs clamped so exp stays finite; no EXEC
// divergence, ~8 VALU ops each instead of OCML tanh's divergent path).
__device__ __forceinline__ float ftanh(float x) {
  float p = fminf(fmaxf(x, -30.f), 30.f);
  float e = __expf(2.f * p);
  return (e - 1.f) * __builtin_amdgcn_rcpf(e + 1.f);
}
__device__ __forceinline__ float fsigmoid(float x) {
  float p = fminf(fmaxf(x, -30.f), 30.f);
  return __builtin_amdgcn_rcpf(1.f + __expf(-p));
}

// A-fragment (16x32 f16) gather from row-major f16 buffer (stride in halves,
// multiple of 8 so rows stay 16B aligned).
// ISA layout: lane L<16 holds row M=L, K = kt*32 + {0..7, 16..23};
//             lane L>=16 holds row M=L-16, K = kt*32 + {8..15, 24..31}.
__device__ __forceinline__ v16h ldA(const _Float16* base, int stride, int kt) {
  const int lane = threadIdx.x & 31;
  const int r = lane & 15, g2 = lane >> 4;
  const _Float16* row = base + r * stride + kt * 32 + g2 * 8;
  v8h lo = *(const v8h*)(row);
  v8h hi = *(const v8h*)(row + 16);
  v16h a;
#pragma unroll
  for (int e = 0; e < 8; ++e) { a[e] = lo[e]; a[8 + e] = hi[e]; }
  return a;
}

// B-fragment (32x16 f16) load from pre-packed global weights:
// packed[((kt*NT + nt)*32 + lane)*16 + e], 32B contiguous per lane.
__device__ __forceinline__ v16h ldB(const _Float16* packed, int kt, int nt, int NT) {
  const int lane = threadIdx.x & 31;
  return *(const v16h*)(packed + ((size_t)(kt * NT + nt) * 32 + lane) * 16);
}

// B-fragment packing index: lane = ((k%16)/8)*16 + (n%16), e = (k%8) + ((k%32)/16)*8
__device__ __forceinline__ size_t packed_idx(int k, int n, int NT) {
  return ((((size_t)(k >> 5) * NT + (n >> 4)) * 32) +
          (((k >> 3) & 1) * 16 + (n & 15))) * 16 + ((k & 7) + (((k >> 4) & 1) << 3));
}

// ---------------------------------------------------------------------------
// Weight packing kernels (fp32 -> f16 B-fragment layout)
// ---------------------------------------------------------------------------
__global__ void k_pack_wbb(const float* __restrict__ Wbb, _Float16* __restrict__ dst) {
  int i = blockIdx.x * blockDim.x + threadIdx.x;       // over V*129*128
  if (i >= 8 * 129 * 128) return;
  int n = i & 127, rv = i >> 7;
  int r = rv % 129, v = rv / 129;
  if (r == 0) return;                                   // row 0 -> rank-1 term
  dst[(size_t)v * 16384 + packed_idx(r - 1, n, 8)] = (_Float16)Wbb[i];
}

__global__ void k_pack_wcat(const float* __restrict__ f1, const float* __restrict__ f2,
                            const float* __restrict__ ta, const float* __restrict__ tb,
                            _Float16* __restrict__ dst) {
  int i = blockIdx.x * blockDim.x + threadIdx.x;       // over V*128*512
  if (i >= 8 * 128 * 512) return;
  int n512 = i & 511, kv = i >> 9;
  int k = kv & 127, v = kv >> 7;
  int which = n512 >> 7, u = n512 & 127;
  const float* s = (which == 0) ? f1 : (which == 1) ? f2 : (which == 2) ? ta : tb;
  dst[(size_t)v * 65536 + packed_idx(k, n512, 32)] =
      (_Float16)s[(size_t)(v * 128 + k) * 128 + u];
}

__global__ void k_pack_wfc1(const float* __restrict__ W, _Float16* __restrict__ dst) {
  int i = blockIdx.x * blockDim.x + threadIdx.x;       // over 1024*1024
  if (i >= 1024 * 1024) return;
  int k = i >> 10, n = i & 1023;
  dst[packed_idx(k, n, 64)] = (_Float16)W[i];
}

// ---------------------------------------------------------------------------
// Recurrent CfC kernel. Block = 128 threads = 4 waves; each wave owns a
// 16-row batch strip for all T steps (fully wave-private -> no barriers in
// the T loop). grid = (B/64, V). Tile loops are kept rolled (unroll 1) so
// the B-fragment loads stay loop-variant: no LICM hoisting, no spills;
// weights stream from L2 (~160KB/variable, fully resident).
// ---------------------------------------------------------------------------
__global__ void __launch_bounds__(128) k_cfc(
    const float* __restrict__ x,        // [V,B,T]
    const float* __restrict__ Wbb_raw,  // [V,129,128] (row 0 -> rank-1 term)
    const _Float16* __restrict__ Wbbp,  // packed [V][4kt][8nt][32][16]
    const _Float16* __restrict__ Wcatp, // packed [V][4kt][32nt][32][16]
    const float* __restrict__ bbb, const float* __restrict__ bf1,
    const float* __restrict__ bf2, const float* __restrict__ bta,
    const float* __restrict__ btb,
    _Float16* __restrict__ feats)       // [B,1024] f16
{
  const int v = blockIdx.y, b0 = blockIdx.x * 64;
  const int tid = threadIdx.x, lane = tid & 31, wave = tid >> 5;
  const int col = lane & 15, g2 = lane >> 4, r0 = g2 * 8;

  __shared__ __align__(16) _Float16 hsh[64 * 136];   // h state, f16
  __shared__ __align__(16) _Float16 zsh[64 * 136];   // backbone act z, f16
  __shared__ __align__(16) _Float16 xs[64 * 128];    // x tile [row][t]
  __shared__ float w0s[128], bbbs[128], bf1s[128], bf2s[128], bts[128];

  for (int i = tid; i < 64 * 128; i += 128) {
    int r = i >> 7, t = i & 127;
    xs[i] = (_Float16)x[((size_t)(v * 256 + b0 + r)) * 128 + t];
  }
  if (tid < 128) {
    w0s[tid]  = Wbb_raw[(size_t)v * 129 * 128 + tid];
    bbbs[tid] = bbb[v * 128 + tid];
    bf1s[tid] = bf1[v * 128 + tid];
    bf2s[tid] = bf2[v * 128 + tid];
    bts[tid]  = bta[v * 128 + tid] + btb[v * 128 + tid];
  }
  _Float16* hrow = hsh + wave * 16 * 136;
  _Float16* zrow = zsh + wave * 16 * 136;
  for (int i = lane; i < 16 * 136; i += 32) hrow[i] = (_Float16)0.f;   // h0 = 0
  __syncthreads();

  const _Float16* wbbv  = Wbbp  + (size_t)v * 16384;
  const _Float16* wcatv = Wcatp + (size_t)v * 65536;
  const int wrow0 = wave * 16;

  for (int t = 0; t < 128; ++t) {
    // ---- GEMM1: z_pre = h @ Wbb[1:] + x*w0 + b ; z = lecun_tanh ----
    v16h ah[4];
#pragma unroll
    for (int kt = 0; kt < 4; ++kt) ah[kt] = ldA(hrow, 136, kt);
    float xr[8];
#pragma unroll
    for (int g = 0; g < 8; ++g) xr[g] = (float)xs[(wrow0 + r0 + g) * 128 + t];

#pragma unroll 1
    for (int nt = 0; nt < 8; ++nt) {
      v16h bq0 = ldB(wbbv, 0, nt, 8);
      v16h bq1 = ldB(wbbv, 1, nt, 8);
      v16h bq2 = ldB(wbbv, 2, nt, 8);
      v16h bq3 = ldB(wbbv, 3, nt, 8);
      v8f acc = {};
      acc = wmma32(ah[0], bq0, acc);
      acc = wmma32(ah[1], bq1, acc);
      acc = wmma32(ah[2], bq2, acc);
      acc = wmma32(ah[3], bq3, acc);
      const int cn = nt * 16 + col;
      const float w0v = w0s[cn], bb = bbbs[cn];
#pragma unroll
      for (int g = 0; g < 8; ++g) {
        float pre = acc[g] + xr[g] * w0v + bb;
        zrow[(r0 + g) * 136 + cn] = (_Float16)(1.7159f * ftanh(0.666f * pre));
      }
    }

    // ---- GEMM2: [ff1|ff2|ta|tb] = z @ Wcat, gate, h update ----
    v16h az[4];
#pragma unroll
    for (int kt = 0; kt < 4; ++kt) az[kt] = ldA(zrow, 136, kt);

#pragma unroll 1
    for (int ut = 0; ut < 8; ++ut) {
      v8f a1 = {}, a2 = {}, a3 = {}, a4 = {};
#pragma unroll
      for (int kt = 0; kt < 4; ++kt) {
        a1 = wmma32(az[kt], ldB(wcatv, kt, ut,      32), a1);
        a2 = wmma32(az[kt], ldB(wcatv, kt, 8 + ut,  32), a2);
        a3 = wmma32(az[kt], ldB(wcatv, kt, 16 + ut, 32), a3);
        a4 = wmma32(az[kt], ldB(wcatv, kt, 24 + ut, 32), a4);
      }
      const int cu = ut * 16 + col;
      const float bb1 = bf1s[cu], bb2 = bf2s[cu], bbt = bts[cu];
#pragma unroll
      for (int g = 0; g < 8; ++g) {
        float f1 = ftanh(a1[g] + bb1);
        float f2 = ftanh(a2[g] + bb2);
        float ti = fsigmoid(a3[g] + a4[g] + bbt);
        hrow[(r0 + g) * 136 + cu] = (_Float16)(f1 + ti * (f2 - f1));
      }
    }
  }
  __syncthreads();

  for (int i = tid; i < 64 * 128; i += 128) {
    int r = i >> 7, u = i & 127;
    feats[(size_t)(b0 + r) * 1024 + v * 128 + u] = hsh[r * 136 + u];
  }
}

// ---------------------------------------------------------------------------
// Classifier: relu(feats @ Wfc1 + b1) @ Wfc2 + b2.
// Block = 256 threads (8 waves) per 16-row M-tile; wave w covers N-tiles
// w*8..w*8+7 (N=1024). grid = 16.
// ---------------------------------------------------------------------------
__global__ void __launch_bounds__(256) k_cls(
    const _Float16* __restrict__ feats, const _Float16* __restrict__ Wfc1p,
    const float* __restrict__ bfc1, const float* __restrict__ Wfc2,
    const float* __restrict__ bfc2, float* __restrict__ out)
{
  const int m0 = blockIdx.x * 16;
  const int tid = threadIdx.x, lane = tid & 31, wave = tid >> 5;
  const int col = lane & 15, g2 = lane >> 4, r0 = g2 * 8;

  __shared__ __align__(16) _Float16 ast[16 * 1032];   // A-strip, reused for relu(h)

  for (int i = tid; i < 16 * 1024; i += 256) {
    int r = i >> 10, u = i & 1023;
    ast[r * 1032 + u] = feats[(size_t)(m0 + r) * 1024 + u];
  }
  __syncthreads();

  v8f acc[8];
#pragma unroll
  for (int j = 0; j < 8; ++j) acc[j] = (v8f){};
#pragma unroll 1
  for (int kt = 0; kt < 32; ++kt) {
    v16h a = ldA(ast, 1032, kt);
#pragma unroll
    for (int j = 0; j < 8; ++j)
      acc[j] = wmma32(a, ldB(Wfc1p, kt, wave * 8 + j, 64), acc[j]);
  }
  __syncthreads();                       // all reads of ast done before reuse

#pragma unroll
  for (int j = 0; j < 8; ++j) {
    const int cn = (wave * 8 + j) * 16 + col;
    const float bb = bfc1[cn];
#pragma unroll
    for (int g = 0; g < 8; ++g) {
      float hv = acc[j][g] + bb;
      ast[(r0 + g) * 1032 + cn] = (_Float16)(hv > 0.f ? hv : 0.f);
    }
  }
  __syncthreads();

  if (tid < 160) {                       // 16 rows x 10 classes
    int r = tid / 10, c = tid % 10;
    float sum = bfc2[c];
    for (int k = 0; k < 1024; ++k)
      sum += (float)ast[r * 1032 + k] * Wfc2[k * 10 + c];
    out[(size_t)(m0 + r) * 10 + c] = sum;
  }
}

// ---------------------------------------------------------------------------
// Launch. Workspace layout (bytes):
//   [0,         262144)  packed Wbb   f16
//   [262144,   1310720)  packed Wcat  f16
//   [1310720,  3407872)  packed Wfc1  f16
//   [3407872,  3932160)  feats        f16
// ---------------------------------------------------------------------------
extern "C" void kernel_launch(void* const* d_in, const int* in_sizes, int n_in,
                              void* d_out, int out_size, void* d_ws, size_t ws_size,
                              hipStream_t stream) {
  const float* x    = (const float*)d_in[0];
  const float* Wbb  = (const float*)d_in[1];
  const float* bbb  = (const float*)d_in[2];
  const float* Wf1  = (const float*)d_in[3];
  const float* bf1  = (const float*)d_in[4];
  const float* Wf2  = (const float*)d_in[5];
  const float* bf2  = (const float*)d_in[6];
  const float* Wta  = (const float*)d_in[7];
  const float* bta  = (const float*)d_in[8];
  const float* Wtb  = (const float*)d_in[9];
  const float* btb  = (const float*)d_in[10];
  const float* Wfc1 = (const float*)d_in[11];
  const float* bfc1 = (const float*)d_in[12];
  const float* Wfc2 = (const float*)d_in[13];
  const float* bfc2 = (const float*)d_in[14];

  char* ws = (char*)d_ws;
  _Float16* wbbp  = (_Float16*)(ws + 0);
  _Float16* wcatp = (_Float16*)(ws + 262144);
  _Float16* wfc1p = (_Float16*)(ws + 1310720);
  _Float16* feats = (_Float16*)(ws + 3407872);

  k_pack_wbb <<<(8 * 129 * 128 + 255) / 256, 256, 0, stream>>>(Wbb, wbbp);
  k_pack_wcat<<<(8 * 128 * 512 + 255) / 256, 256, 0, stream>>>(Wf1, Wf2, Wta, Wtb, wcatp);
  k_pack_wfc1<<<(1024 * 1024) / 256,          256, 0, stream>>>(Wfc1, wfc1p);

  k_cfc<<<dim3(4, 8), 128, 0, stream>>>(x, Wbb, wbbp, wcatp,
                                        bbb, bf1, bf2, bta, btb, feats);
  k_cls<<<16, 256, 0, stream>>>(feats, wfc1p, bfc1, Wfc2, bfc2, (float*)d_out);
}
